// OctreeDeconvGnRelu_25649544692150
// MI455X (gfx1250) — matrix operand order repaired and checked by reading, not compiled, not run on hardware
//
#include <hip/hip_runtime.h>

typedef __attribute__((ext_vector_type(16))) _Float16 v16h;
typedef __attribute__((ext_vector_type(8)))  _Float16 v8h;
typedef __attribute__((ext_vector_type(4)))  _Float16 v4h;
typedef __attribute__((ext_vector_type(8)))  float    v8f;

#define KTAP 27
#define CIN  32
#define COUT 32
#define GGRP 8
#define NBATCH 8
#define EPSV 1e-5f
#define PACKN (KTAP * 2 * 32 * 16)   // halves in packed weight buffer

// ---------------------------------------------------------------------------
// Repack weights[k][cin][cout] (f32) into WMMA B-fragment layout (f16).
// Fragment for (tap k, cout-tile t): lane L holds v16h with
//   B[i] = W[k][ (L>=16?8:0)+(i&7)+(i>=8?16:0) ][ t*16 + (L&15) ]
// ---------------------------------------------------------------------------
__global__ void pack_weights(const float* __restrict__ w, _Float16* __restrict__ wpack) {
    int j = blockIdx.x * blockDim.x + threadIdx.x;
    if (j >= PACKN) return;
    int i    = j & 15;
    int lane = (j >> 4) & 31;
    int t    = (j >> 9) & 1;
    int k    = j >> 10;
    int kk   = ((lane >> 4) << 3) + (i & 7) + ((i >> 3) << 4);        // K row within 0..31
    int co   = (t << 4) + (lane & 15);        // cout column
    wpack[j] = (_Float16)w[((size_t)(k * CIN + kk)) * COUT + co];
}

// ---------------------------------------------------------------------------
// One-time f32 -> f16 conversion of the node features (streaming pass).
// ---------------------------------------------------------------------------
__global__ __launch_bounds__(256)
void convert_data_f16(const float* __restrict__ src, _Float16* __restrict__ dst, int total4) {
    int i = blockIdx.x * blockDim.x + threadIdx.x;
    if (i >= total4) return;
    float4 v = ((const float4*)src)[i];
    v4h h;
    h[0] = (_Float16)v.x; h[1] = (_Float16)v.y;
    h[2] = (_Float16)v.z; h[3] = (_Float16)v.w;
    ((v4h*)dst)[i] = h;
}

// ---------------------------------------------------------------------------
// Deconv (f16-gather): one wave per 32-node super-tile (two 16-row M tiles).
// B fragments are reused across both M tiles: per tap, 2 A loads + 2 B loads
// feed 4 independent v_wmma_f32_16x16x32_f16 (32 B of loads per WMMA).
// ---------------------------------------------------------------------------
__global__ __launch_bounds__(256)
void deconv_wmma_h(const _Float16* __restrict__ data16, const _Float16* __restrict__ wpack,
                   const int* __restrict__ neigh, float* __restrict__ out, int n) {
    const int wave = threadIdx.x >> 5;
    const int lane = threadIdx.x & 31;
    const int st   = blockIdx.x * 8 + wave;       // super-tile (32 nodes)
    const int numST = (n + 31) >> 5;
    if (st >= numST) return;

    const int half = lane >> 4;               // 0: K chunks {0-7,16-23}; 1: {8-15,24-31}
    const int m    = lane & 15;               // A row M = lane % 16
    const int base = st * 32;
    int node0 = base + m;                     // M tile 0
    int node1 = base + 16 + m;                // M tile 1
    if (node0 >= n) node0 = n - 1;            // clamp (keeps EXEC all-1 for WMMA)
    if (node1 >= n) node1 = n - 1;
    const int coff = half * 8;

    v8f acc00 = {}, acc01 = {};               // node-tile 0, cout tiles 0/1
    v8f acc10 = {}, acc11 = {};               // node-tile 1, cout tiles 0/1
    const int nbase0 = node0 * KTAP;
    const int nbase1 = node1 * KTAP;

    for (int k = 0; k < KTAP; ++k) {
        int r0 = neigh[nbase0 + k];
        int r1 = neigh[nbase1 + k];
        const _Float16* s0 = data16 + (size_t)r0 * CIN + coff;
        const _Float16* s1 = data16 + (size_t)r1 * CIN + coff;
        v8h a0lo = *(const v8h*)(s0);         // K = coff+0..7
        v8h a0hi = *(const v8h*)(s0 + 16);    // K = coff+16..23
        v8h a1lo = *(const v8h*)(s1);
        v8h a1hi = *(const v8h*)(s1 + 16);
        v16h A0 = __builtin_shufflevector(a0lo, a0hi,
                                          0, 1, 2, 3, 4, 5, 6, 7,
                                          8, 9, 10, 11, 12, 13, 14, 15);
        v16h A1 = __builtin_shufflevector(a1lo, a1hi,
                                          0, 1, 2, 3, 4, 5, 6, 7,
                                          8, 9, 10, 11, 12, 13, 14, 15);

        const v16h B0 = *(const v16h*)(wpack + ((size_t)(k * 2 + 0) * 32 + lane) * 16);
        const v16h B1 = *(const v16h*)(wpack + ((size_t)(k * 2 + 1) * 32 + lane) * 16);

        acc00 = __builtin_amdgcn_wmma_f32_16x16x32_f16(false, A0, false, B0,
                                                       (short)0, acc00, false, false);
        acc01 = __builtin_amdgcn_wmma_f32_16x16x32_f16(false, A0, false, B1,
                                                       (short)0, acc01, false, false);
        acc10 = __builtin_amdgcn_wmma_f32_16x16x32_f16(false, A1, false, B0,
                                                       (short)0, acc10, false, false);
        acc11 = __builtin_amdgcn_wmma_f32_16x16x32_f16(false, A1, false, B1,
                                                       (short)0, acc11, false, false);
    }

    // C/D layout: lanes 0-15 -> N=lane, M=vgpr; lanes 16-31 -> N=lane-16, M=8+vgpr
    const int ncol = lane & 15;
#pragma unroll
    for (int r2 = 0; r2 < 8; ++r2) {
        int M   = half * 8 + r2;
        int on0 = base + M;
        int on1 = base + 16 + M;
        if (on0 < n) {
            out[(size_t)on0 * COUT + ncol]      = acc00[r2];
            out[(size_t)on0 * COUT + 16 + ncol] = acc01[r2];
        }
        if (on1 < n) {
            out[(size_t)on1 * COUT + ncol]      = acc10[r2];
            out[(size_t)on1 * COUT + 16 + ncol] = acc11[r2];
        }
    }
}

// ---------------------------------------------------------------------------
// Deconv (f32-gather fallback, used if workspace can't hold f16 data copy).
// ---------------------------------------------------------------------------
__global__ __launch_bounds__(256)
void deconv_wmma_f(const float* __restrict__ data, const _Float16* __restrict__ wpack,
                   const int* __restrict__ neigh, float* __restrict__ out, int n) {
    const int wave = threadIdx.x >> 5;
    const int lane = threadIdx.x & 31;
    const int tile = blockIdx.x * 8 + wave;
    const int numTiles = (n + 15) >> 4;
    if (tile >= numTiles) return;

    const int half = lane >> 4;
    const int m    = lane & 15;
    int node = tile * 16 + m;
    if (node >= n) node = n - 1;
    const int coff = half * 8;

    v8f acc0 = {};
    v8f acc1 = {};
    const int nbase = node * KTAP;

    for (int k = 0; k < KTAP; ++k) {
        int r = neigh[nbase + k];
        const float* src = data + (size_t)r * CIN + coff;
        float4 a0 = *(const float4*)(src + 0);
        float4 a1 = *(const float4*)(src + 4);
        float4 a2 = *(const float4*)(src + 16);
        float4 a3 = *(const float4*)(src + 20);
        v16h A;
        A[0]  = (_Float16)a0.x; A[1]  = (_Float16)a0.y;
        A[2]  = (_Float16)a0.z; A[3]  = (_Float16)a0.w;
        A[4]  = (_Float16)a1.x; A[5]  = (_Float16)a1.y;
        A[6]  = (_Float16)a1.z; A[7]  = (_Float16)a1.w;
        A[8]  = (_Float16)a2.x; A[9]  = (_Float16)a2.y;
        A[10] = (_Float16)a2.z; A[11] = (_Float16)a2.w;
        A[12] = (_Float16)a3.x; A[13] = (_Float16)a3.y;
        A[14] = (_Float16)a3.z; A[15] = (_Float16)a3.w;

        const v16h B0 = *(const v16h*)(wpack + ((size_t)(k * 2 + 0) * 32 + lane) * 16);
        const v16h B1 = *(const v16h*)(wpack + ((size_t)(k * 2 + 1) * 32 + lane) * 16);

        acc0 = __builtin_amdgcn_wmma_f32_16x16x32_f16(false, A, false, B0,
                                                      (short)0, acc0, false, false);
        acc1 = __builtin_amdgcn_wmma_f32_16x16x32_f16(false, A, false, B1,
                                                      (short)0, acc1, false, false);
    }

    const int ncol = lane & 15;
#pragma unroll
    for (int r2 = 0; r2 < 8; ++r2) {
        int M  = half * 8 + r2;
        int on = tile * 16 + M;
        if (on < n) {
            out[(size_t)on * COUT + ncol]      = acc0[r2];
            out[(size_t)on * COUT + 16 + ncol] = acc1[r2];
        }
    }
}

// ---------------------------------------------------------------------------
// Group-norm statistics (one-pass sum / sumsq per (batch, channel) + counts).
// ---------------------------------------------------------------------------
__global__ void zero_stats(float* p, int cnt) {
    int i = blockIdx.x * blockDim.x + threadIdx.x;
    if (i < cnt) p[i] = 0.0f;
}

__global__ __launch_bounds__(256)
void reduce_stats(const float* __restrict__ h, const int* __restrict__ batch_id,
                  float* __restrict__ gsum, float* __restrict__ gsq,
                  float* __restrict__ gcnt, int n) {
    __shared__ float s_sum[NBATCH * COUT];
    __shared__ float s_sq[NBATCH * COUT];
    __shared__ float s_cnt[NBATCH];
    for (int i = threadIdx.x; i < NBATCH * COUT; i += blockDim.x) {
        s_sum[i] = 0.0f; s_sq[i] = 0.0f;
    }
    if (threadIdx.x < NBATCH) s_cnt[threadIdx.x] = 0.0f;
    __syncthreads();

    const int c = threadIdx.x & 31;           // one wave covers one node's 32 channels
    const int rowInBlk = threadIdx.x >> 5;    // 8 nodes per block per iter
    for (int node = blockIdx.x * 8 + rowInBlk; node < n; node += gridDim.x * 8) {
        int b = batch_id[node];
        float v = h[(size_t)node * COUT + c];
        atomicAdd(&s_sum[b * COUT + c], v);
        atomicAdd(&s_sq[b * COUT + c], v * v);
        if (c == 0) atomicAdd(&s_cnt[b], 1.0f);
    }
    __syncthreads();

    for (int i = threadIdx.x; i < NBATCH * COUT; i += blockDim.x) {
        atomicAdd(&gsum[i], s_sum[i]);
        atomicAdd(&gsq[i], s_sq[i]);
    }
    if (threadIdx.x < NBATCH) atomicAdd(&gcnt[threadIdx.x], s_cnt[threadIdx.x]);
}

// Per-(b,c) mean and scale = inv_std * gamma, matching reference:
//   inv_cnt = 1/(n_b*(C/G)+eps); mean_g = sum_g*inv_cnt; var_g = sum(hc^2)*inv_cnt
__global__ void finalize_stats(const float* __restrict__ gsum, const float* __restrict__ gsq,
                               const float* __restrict__ gcnt, const float* __restrict__ gamma,
                               float* __restrict__ meanArr, float* __restrict__ scaleArr) {
    int t = threadIdx.x;
    if (t >= NBATCH * COUT) return;
    int b = t >> 5;
    int c = t & 31;
    int g = c >> 2;                            // Cg = COUT/GGRP = 4
    int base = b * COUT + (g << 2);
    float sum_g = gsum[base] + gsum[base + 1] + gsum[base + 2] + gsum[base + 3];
    float sq_g  = gsq[base]  + gsq[base + 1]  + gsq[base + 2]  + gsq[base + 3];
    float cntf  = gcnt[b] * (float)(COUT / GGRP);
    float inv   = 1.0f / (cntf + EPSV);
    float mean  = sum_g * inv;
    float varsum = sq_g - 2.0f * mean * sum_g + cntf * mean * mean; // = sum (h-mean)^2
    float var   = varsum * inv;
    float istd  = rsqrtf(var + EPSV);
    meanArr[t]  = mean;
    scaleArr[t] = istd * gamma[c];
}

__global__ __launch_bounds__(256)
void apply_gn_relu(float* __restrict__ h, const int* __restrict__ batch_id,
                   const float* __restrict__ meanArr, const float* __restrict__ scaleArr,
                   const float* __restrict__ beta, int n) {
    int idx = blockIdx.x * blockDim.x + threadIdx.x;
    if (idx >= n * COUT) return;
    int node = idx >> 5;
    int c    = idx & 31;
    int b    = batch_id[node];
    float v  = (h[idx] - meanArr[b * COUT + c]) * scaleArr[b * COUT + c] + beta[c];
    h[idx]   = v > 0.0f ? v : 0.0f;
}

// ---------------------------------------------------------------------------
extern "C" void kernel_launch(void* const* d_in, const int* in_sizes, int n_in,
                              void* d_out, int out_size, void* d_ws, size_t ws_size,
                              hipStream_t stream) {
    const float* data     = (const float*)d_in[0];
    const float* weights  = (const float*)d_in[1];
    const float* gamma    = (const float*)d_in[2];
    const float* beta     = (const float*)d_in[3];
    const int*   neigh    = (const int*)d_in[4];
    const int*   batch_id = (const int*)d_in[5];
    const int    n        = in_sizes[0] / CIN;
    float*       out      = (float*)d_out;

    // Preferred workspace layout: [data16 f16 | wpack f16 | stats f32]
    size_t data16Bytes  = ((size_t)n * CIN * sizeof(_Float16) + 255) & ~(size_t)255;
    size_t wpackBytes   = ((size_t)PACKN * sizeof(_Float16) + 255) & ~(size_t)255;
    size_t statsBytes   = 1032 * sizeof(float);
    bool   useF16Data   = ws_size >= data16Bytes + wpackBytes + statsBytes;

    _Float16* data16;
    _Float16* wpack;
    float*    stats;
    if (useF16Data) {
        data16 = (_Float16*)d_ws;
        wpack  = (_Float16*)((char*)d_ws + data16Bytes);
        stats  = (float*)((char*)d_ws + data16Bytes + wpackBytes);
    } else {
        data16 = nullptr;
        wpack  = (_Float16*)d_ws;
        stats  = (float*)((char*)d_ws + wpackBytes);
    }
    float* s_sum  = stats;                 // 256
    float* s_sq   = stats + 256;           // 256
    float* s_cnt  = stats + 512;           // 8
    float* s_mean = stats + 520;           // 256
    float* s_scal = stats + 776;           // 256

    pack_weights<<<(PACKN + 255) / 256, 256, 0, stream>>>(weights, wpack);
    zero_stats<<<(520 + 255) / 256, 256, 0, stream>>>(stats, 520);

    if (useF16Data) {
        int total4 = (n * CIN) / 4;
        convert_data_f16<<<(total4 + 255) / 256, 256, 0, stream>>>(data, data16, total4);
        int numST = (n + 31) / 32;
        deconv_wmma_h<<<(numST + 7) / 8, 256, 0, stream>>>(data16, wpack, neigh, out, n);
    } else {
        int numTiles = (n + 15) / 16;
        deconv_wmma_f<<<(numTiles + 7) / 8, 256, 0, stream>>>(data, wpack, neigh, out, n);
    }

    reduce_stats<<<512, 256, 0, stream>>>(out, batch_id, s_sum, s_sq, s_cnt, n);
    finalize_stats<<<1, 256, 0, stream>>>(s_sum, s_sq, s_cnt, gamma, s_mean, s_scal);
    apply_gn_relu<<<((size_t)n * COUT + 255) / 256, 256, 0, stream>>>(out, batch_id,
                                                                      s_mean, s_scal, beta, n);
}
